// RNN_49941879718205
// MI455X (gfx1250) — matrix operand order repaired and checked by reading, not compiled, not run on hardware
//
#include <hip/hip_runtime.h>
#include <hip/hip_bf16.h>

// ---------------------------------------------------------------------------
// Persistent-LSTM for MI455X (gfx1250).
//   - 64 workgroups, one per WGP, each owning 16 h-columns (=> 64 gate cols).
//   - Wi/Wh slices converted to bf16 and pre-swizzled into WMMA B-fragment
//     layout in LDS once; reused across all 512 timesteps.
//   - Per step: stage x_t (f32->bf16, VALU) + h (bf16, async global->LDS DMA)
//     into LDS, 48x v_wmma_f32_16x16x32_bf16 per wave, gate pointwise math,
//     publish h via device-wide barrier.
// ---------------------------------------------------------------------------

typedef __attribute__((ext_vector_type(16))) __bf16 v16bf;
typedef __attribute__((ext_vector_type(8)))  __bf16 v8bf;
typedef __attribute__((ext_vector_type(8)))  float  v8f;

#define B_    32
#define T_    512
#define D_    512
#define H_    1024
#define G4H   4096
#define NWG   64          // 1024 h-cols / 16 per WG
#define KX    16          // k-chunks (of 32) from x
#define KH    32          // k-chunks (of 32) from h
#define APITCH 1544       // 1536 + 8 ushort pad (bank-conflict avoidance)

// dynamic-LDS layout (bytes)
#define OFF_A    0                         // ushort[32][1544]      = 98816
#define OFF_WI   98816                     // ushort[16*4*32*16]    = 65536
#define OFF_WH   164352                    // ushort[32*4*32*16]    = 131072
#define OFF_Z    295424                    // float[4][32][16]      = 8192
#define OFF_C    303616                    // float[32][16]         = 2048
#define OFF_B    305664                    // float[64]             = 256
#define SMEM_BYTES 305920                  // ~299 KB < 320 KB WGP LDS

__device__ __forceinline__ unsigned short f2bf(float f) {
  unsigned u = __builtin_bit_cast(unsigned, f);
  u += 0x7FFFu + ((u >> 16) & 1u);         // round-to-nearest-even
  return (unsigned short)(u >> 16);
}

__device__ __forceinline__ float sigmoidf_(float v) {
  return 1.0f / (1.0f + __expf(-v));
}

union V16U { v16bf v; v8bf h[2]; };

extern "C" __global__ void lstm_init(const float* __restrict__ h0,
                                     unsigned short* __restrict__ hbuf0,
                                     unsigned* __restrict__ gctr) {
  int i = blockIdx.x * blockDim.x + threadIdx.x;
  if (i == 0) *gctr = 0u;
  if (i < B_ * H_) hbuf0[i] = f2bf(h0[i]);
}

extern "C" __global__ __launch_bounds__(256, 1)
void lstm_persistent(const float* __restrict__ c0,
                     const float* __restrict__ x,
                     const float* __restrict__ Wi,
                     const float* __restrict__ Wh,
                     const float* __restrict__ bias,
                     unsigned short* __restrict__ hbuf0,
                     unsigned short* __restrict__ hbuf1,
                     unsigned* __restrict__ gctr,
                     float* __restrict__ out) {
  extern __shared__ char smem[];
  unsigned short* As  = (unsigned short*)(smem + OFF_A);
  unsigned short* WiF = (unsigned short*)(smem + OFF_WI);
  unsigned short* WhF = (unsigned short*)(smem + OFF_WH);
  float* zb = (float*)(smem + OFF_Z);
  float* cb = (float*)(smem + OFF_C);
  float* bb = (float*)(smem + OFF_B);

  const int tid = threadIdx.x;
  const int wg  = blockIdx.x;        // owns h-cols [wg*16, wg*16+16)
  const int hc0 = wg * 16;

  // ---- one-time setup: pre-swizzle weight slices into WMMA B-fragment form.
  // B-fragment (32x16 bf16): lane L holds column n = L&15; element j holds
  // K = 16*(L>>4) + j   (contiguous k order chosen here; A loads use the same).
  for (int e = tid; e < KX * 4 * 32 * 16; e += 256) {
    int j = e & 15, lane = (e >> 4) & 31, q = (e >> 9) & 3, kc = e >> 11;
    int k   = kc * 32 + 16 * (lane >> 4) + j;
    int col = q * H_ + hc0 + (lane & 15);
    WiF[e] = f2bf(Wi[(size_t)k * G4H + col]);
  }
  for (int e = tid; e < KH * 4 * 32 * 16; e += 256) {
    int j = e & 15, lane = (e >> 4) & 31, q = (e >> 9) & 3, kc = e >> 11;
    int k   = kc * 32 + 16 * (lane >> 4) + j;
    int col = q * H_ + hc0 + (lane & 15);
    WhF[e] = f2bf(Wh[(size_t)k * G4H + col]);
  }
  for (int e = tid; e < 512; e += 256)
    cb[e] = c0[(e >> 4) * H_ + hc0 + (e & 15)];
  if (tid < 64) bb[tid] = bias[(tid >> 4) * H_ + hc0 + (tid & 15)];

  const int lane = tid & 31;
  const int wv   = tid >> 5;   // wave 0..7
  const int mt   = wv >> 2;    // M-tile: batch rows [16*mt, 16*mt+16)
  const int q    = wv & 3;     // gate: 0=i 1=f 2=g 3=o
  const int half = lane >> 4;
  const int l15  = lane & 15;

  float* ys = out + 2 * B_ * H_;

  __syncthreads();

  for (int t = 0; t < T_; ++t) {
    const unsigned short* hprev = (t & 1) ? hbuf1 : hbuf0;
    unsigned short*       hnext = (t & 1) ? hbuf0 : hbuf1;

    // ---- stage h (raw bf16 bytes) into A[:, 512:1536] with the CDNA5
    // async global->LDS DMA path (no VGPR staging; tracked by ASYNCcnt).
    // LDS destination address = low 32 bits of the flat pointer (ISA 10.2:
    // flat addresses in the LDS aperture keep the allocation-relative
    // offset in addr[31:0]).
    for (int idx = tid; idx < 32 * 128; idx += 256) {
      int b = idx >> 7, u = idx & 127;
      unsigned long long ga =
          (unsigned long long)(const void*)(hprev + b * H_ + u * 8);
      unsigned ldsoff =
          (unsigned)(unsigned long long)(void*)(As + b * APITCH + 512 + u * 8);
      asm volatile("global_load_async_to_lds_b128 %0, %1, off"
                   :: "v"(ldsoff), "v"(ga)
                   : "memory");
    }

    // ---- stage x_t (f32 -> bf16) into A[:, 0:512] (needs conversion, so
    // it rides the VALU path and overlaps the async h copy above)
    for (int idx = tid; idx < 32 * 64; idx += 256) {
      int b = idx >> 6, u = idx & 63;
      const float4* src = (const float4*)(x + ((size_t)b * T_ + t) * D_ + u * 8);
      float4 a0 = src[0], a1 = src[1];
      uint4 o;
      o.x = (unsigned)f2bf(a0.x) | ((unsigned)f2bf(a0.y) << 16);
      o.y = (unsigned)f2bf(a0.z) | ((unsigned)f2bf(a0.w) << 16);
      o.z = (unsigned)f2bf(a1.x) | ((unsigned)f2bf(a1.y) << 16);
      o.w = (unsigned)f2bf(a1.z) | ((unsigned)f2bf(a1.w) << 16);
      *(uint4*)(As + b * APITCH + u * 8) = o;
    }
    // prefetch next timestep's x rows (global_prefetch_b8)
    if (t + 1 < T_ && tid < 32)
      __builtin_prefetch(x + ((size_t)tid * T_ + (t + 1)) * D_, 0, 1);

    // drain this thread's async LDS copies, then rendezvous
    asm volatile("s_wait_asynccnt 0x0" ::: "memory");
    __syncthreads();

    // ---- GEMM: this wave's 16x16 tile of z = [x_t|h] * [Wi|Wh] slice
    // A-fragment (16x32 bf16): lane L row = L&15; k-run [16*half,+16)
    // (contiguous, matching the B-fragment order).
    v8f acc = {0.f, 0.f, 0.f, 0.f, 0.f, 0.f, 0.f, 0.f};
    const unsigned short* arow = As + (16 * mt + l15) * APITCH + 16 * half;
#pragma unroll 4
    for (int kc = 0; kc < KX; ++kc) {
      V16U a;
      const unsigned short* ap = arow + kc * 32;
      a.h[0] = *(const v8bf*)(ap);
      a.h[1] = *(const v8bf*)(ap + 8);
      v16bf bf = *(const v16bf*)(WiF + (((kc * 4) + q) * 32 + lane) * 16);
      acc = __builtin_amdgcn_wmma_f32_16x16x32_bf16(false, a.v, false, bf,
                                                    (short)0, acc, false, false);
    }
#pragma unroll 4
    for (int kc = 0; kc < KH; ++kc) {
      V16U a;
      const unsigned short* ap = arow + (KX + kc) * 32;
      a.h[0] = *(const v8bf*)(ap);
      a.h[1] = *(const v8bf*)(ap + 8);
      v16bf bf = *(const v16bf*)(WhF + (((kc * 4) + q) * 32 + lane) * 16);
      acc = __builtin_amdgcn_wmma_f32_16x16x32_bf16(false, a.v, false, bf,
                                                    (short)0, acc, false, false);
    }

    // ---- scatter D tile to zbuf: D layout m = v + 8*half (+16*mt), n = l15
#pragma unroll
    for (int v = 0; v < 8; ++v) {
      int m = 16 * mt + 8 * half + v;
      zb[(q * 32 + m) * 16 + l15] = acc[v];
    }
    __syncthreads();

    // ---- gates + state update (2 elements per thread)
    for (int e = tid; e < 512; e += 256) {
      int m = e >> 4, n = e & 15;
      float zi = zb[(0 * 32 + m) * 16 + n] + bb[n];
      float zf = zb[(1 * 32 + m) * 16 + n] + bb[16 + n];
      float zg = zb[(2 * 32 + m) * 16 + n] + bb[32 + n];
      float zo = zb[(3 * 32 + m) * 16 + n] + bb[48 + n];
      float ig = sigmoidf_(zi);
      float fg = sigmoidf_(zf);
      float gg = tanhf(zg);
      float og = sigmoidf_(zo);
      float cn = fg * cb[e] + ig * gg;
      cb[e] = cn;
      float hn = og * tanhf(cn);
      int hcol = hc0 + n;
      ys[((size_t)m * T_ + t) * H_ + hcol] = hn;   // y[b,t,h]
      hnext[m * H_ + hcol] = f2bf(hn);
      if (t == T_ - 1) {
        out[m * H_ + hcol]            = cn;        // c_T
        out[B_ * H_ + m * H_ + hcol]  = hn;        // h_T
      }
    }

    // ---- device-wide barrier (monotonic counter; reset by lstm_init)
    __threadfence();
    __syncthreads();
    if (tid == 0) {
      __hip_atomic_fetch_add(gctr, 1u, __ATOMIC_RELEASE,
                             __HIP_MEMORY_SCOPE_AGENT);
      unsigned target = (unsigned)(NWG * (t + 1));
      while (__hip_atomic_load(gctr, __ATOMIC_ACQUIRE,
                               __HIP_MEMORY_SCOPE_AGENT) < target) {
        __builtin_amdgcn_s_sleep(1);
      }
    }
    __syncthreads();
  }
}

extern "C" void kernel_launch(void* const* d_in, const int* in_sizes, int n_in,
                              void* d_out, int out_size, void* d_ws,
                              size_t ws_size, hipStream_t stream) {
  const float* c0 = (const float*)d_in[0];
  const float* h0 = (const float*)d_in[1];
  const float* x  = (const float*)d_in[2];
  const float* Wi = (const float*)d_in[3];
  const float* Wh = (const float*)d_in[4];
  const float* b  = (const float*)d_in[5];
  float* out = (float*)d_out;

  unsigned* gctr = (unsigned*)d_ws;
  unsigned short* hbuf0 = (unsigned short*)((char*)d_ws + 256);
  unsigned short* hbuf1 = hbuf0 + B_ * H_;

  lstm_init<<<(B_ * H_ + 255) / 256, 256, 0, stream>>>(h0, hbuf0, gctr);
  lstm_persistent<<<NWG, 256, SMEM_BYTES, stream>>>(c0, x, Wi, Wh, b,
                                                    hbuf0, hbuf1, gctr, out);
}
